// MultiHeadAttentionImproved_34299608826115
// MI455X (gfx1250) — matrix operand order repaired and checked by reading, not compiled, not run on hardware
//
#include <hip/hip_runtime.h>
#include <hip/hip_bf16.h>
#include <math.h>
#include <stdint.h>

// ---------------------------------------------------------------------------
// MHA + ALiBi for MI455X (gfx1250): bf16 WMMA + async-LDS staging pipeline.
//   B=2, Sq=Sk=2048, D=1024, H=16, dk=64
// ---------------------------------------------------------------------------

typedef __attribute__((ext_vector_type(16))) __bf16 v16bf;
typedef __attribute__((ext_vector_type(8)))  float  v8f;

union Frag {            // 32 bytes: one 16-half WMMA operand per lane
    uint4 q[2];
    v16bf b;
};

static __device__ inline v8f wmma_bf16(v16bf a, v16bf b, v8f c) {
    // D = A(16x32 bf16) * B(32x16 bf16) + C(16x16 f32)
    return __builtin_amdgcn_wmma_f32_16x16x32_bf16(
        false, a, false, b, (short)0, c, false, false);
}

static __device__ inline unsigned short f2bf(float f) {   // round-to-nearest-even
    unsigned u = __float_as_uint(f);
    unsigned r = u + 0x7FFFu + ((u >> 16) & 1u);
    return (unsigned short)(r >> 16);
}

// Async global->LDS copy, 16B per lane (ASYNCcnt-tracked, gfx1250 path).
// LDS operand = low 32 bits of the flat shared-memory address (ISA 10.2:
// LDS aperture maps addr[31:0] to the wave-relative LDS offset).
static __device__ inline void async_copy_b128(void* lds_dst, const void* gsrc) {
    unsigned lds_off = (unsigned)(uintptr_t)lds_dst;
    asm volatile("global_load_async_to_lds_b128 %0, %1, off"
                 :: "v"(lds_off), "v"(gsrc) : "memory");
}
static __device__ inline void wait_async0() {
    asm volatile("s_wait_asynccnt 0" ::: "memory");
}

// ------------------------------- fp32 -> bf16 ------------------------------
__global__ void cvt_bf16_kernel(const float* __restrict__ src,
                                unsigned short* __restrict__ dst, int n) {
    int i = blockIdx.x * blockDim.x + threadIdx.x;
    if (i < n) dst[i] = f2bf(src[i]);
}

// ------------------------- GEMM: C = A * W^T (+epilogue) -------------------
// A: [M,Kd] bf16 row-major.  W: [N,Kd] bf16 row-major (used transposed).
// Block = 8 waves = 128x64 tile; W tile (64x32 = 4KB) double-buffered in LDS
// via async copies and shared by all 8 waves.
// mode 0/1: out bf16 -> [B,H,2048,64] (Q / K)   mode 2: bf16 -> [B,H,64,2048]
// mode 3  : out f32  -> d_out [M,N] + bias[n]
__global__ void gemm_xwt_kernel(const unsigned short* __restrict__ A,
                                const unsigned short* __restrict__ W,
                                unsigned short* __restrict__ outB,
                                float* __restrict__ outF,
                                const float* __restrict__ bias,
                                int M, int N, int Kd, int mode) {
    __shared__ __align__(16) unsigned short wbuf[2][64 * 32];  // 2 x 4KB

    const int tid  = threadIdx.x;
    const int lane = tid & 31;
    const int wv   = tid >> 5;
    const int half = lane >> 4;
    const int ln   = lane & 15;
    const int m0 = blockIdx.x * 128 + wv * 16;
    const int n0 = blockIdx.y * 64;

    // stage W tile [n0..n0+63] x [kb..kb+31] into wbuf[buf]; 16B per lane.
    auto stageW = [&](int kb, int buf) {
        const unsigned short* g = W + (size_t)(n0 + (tid >> 2)) * Kd + kb + (tid & 3) * 8;
        async_copy_b128(&wbuf[buf][tid * 8], g);
    };

    const v8f zero = {0.f, 0.f, 0.f, 0.f, 0.f, 0.f, 0.f, 0.f};
    v8f acc[4] = {zero, zero, zero, zero};

    stageW(0, 0);

    const unsigned short* arow = A + (size_t)(m0 + ln) * Kd;
    Frag a;                       // A frag for current k-step (pipelined)
    a.q[0] = *(const uint4*)(arow + half * 8);
    a.q[1] = *(const uint4*)(arow + 16 + half * 8);

    int buf = 0;
    for (int kb = 0; kb < Kd; kb += 32) {
        wait_async0();            // own async copies done
        __syncthreads();          // whole tile visible / prev reads finished

        if (kb + 32 < Kd) stageW(kb + 32, buf ^ 1);

        // B frags from LDS: col N=n0+t*16+ln, K at half*8 and 16+half*8
        Frag bfr[4];
#pragma unroll
        for (int t = 0; t < 4; ++t) {
            const uint4* p = (const uint4*)&wbuf[buf][(t * 16 + ln) * 32 + half * 8];
            bfr[t].q[0] = p[0];
            bfr[t].q[1] = p[2];
        }

        Frag an = a;              // prefetch next A while WMMAs run
        if (kb + 32 < Kd) {
            an.q[0] = *(const uint4*)(arow + kb + 32 + half * 8);
            an.q[1] = *(const uint4*)(arow + kb + 48 + half * 8);
        }

#pragma unroll
        for (int t = 0; t < 4; ++t) acc[t] = wmma_bf16(a.b, bfr[t].b, acc[t]);

        a = an;
        buf ^= 1;
    }

    // Epilogue. C layout: vgpr i -> row m0+i+8*half, col n0+t*16+ln.
#pragma unroll
    for (int t = 0; t < 4; ++t) {
#pragma unroll
        for (int i = 0; i < 8; ++i) {
            const int m = m0 + i + 8 * half;
            const int n = n0 + t * 16 + ln;
            const float v = acc[t][i];
            if (mode == 3) {
                outF[(size_t)m * N + n] = v + bias[n];
            } else if (mode == 2) {   // V^T: [B,H,64,2048]
                const int b = m >> 11, s = m & 2047, hh = n >> 6, d = n & 63;
                outB[(((size_t)(b * 16 + hh) * 64 + d) << 11) + s] = f2bf(v);
            } else {                  // Q/K: [B,H,2048,64]
                const int b = m >> 11, s = m & 2047, hh = n >> 6, d = n & 63;
                outB[((((size_t)(b * 16 + hh) << 11) + s) << 6) + d] = f2bf(v);
            }
        }
    }
}

// -------------------- flash attention with ALiBi ---------------------------
// Block = 8 waves sharing one (b,h); each wave owns 16 query rows.
// K / V^T tiles (4KB each per 32-key step) double-buffered in LDS via
// async copies and shared by all 8 waves.
__global__ void attn_alibi_kernel(const unsigned short* __restrict__ Qh,
                                  const unsigned short* __restrict__ Kh,
                                  const unsigned short* __restrict__ Vt,
                                  unsigned short* __restrict__ ctx) {
    __shared__ __align__(16) unsigned short kbuf[2][32 * 64];  // keys x d
    __shared__ __align__(16) unsigned short vbuf[2][64 * 32];  // d x keys
    __shared__ __align__(16) unsigned short pbuf[8][16 * 32];  // P staging / wave

    const int tid  = threadIdx.x;
    const int lane = tid & 31;
    const int wv   = tid >> 5;
    const int half = lane >> 4;
    const int ln   = lane & 15;

    const int qg = blockIdx.x & 15;            // 16 q-groups of 8 waves
    const int h  = (blockIdx.x >> 4) & 15;
    const int b  = blockIdx.x >> 8;
    const int q0 = (qg * 8 + wv) * 16;

    const unsigned short* Qs = Qh + ((size_t)(b * 16 + h) << 11) * 64;
    const unsigned short* Ks = Kh + ((size_t)(b * 16 + h) << 11) * 64;
    const unsigned short* Vs = Vt + ((size_t)(b * 16 + h) * 64 << 11);

    auto stageK = [&](int kb, int buf) {   // rows kb..kb+31, 64 halves each
        const unsigned short* g = Ks + (size_t)(kb + (tid >> 3)) * 64 + (tid & 7) * 8;
        async_copy_b128(&kbuf[buf][tid * 8], g);
    };
    auto stageV = [&](int kb, int buf) {   // rows d=0..63, 32 halves at kb
        const unsigned short* g = Vs + ((size_t)(tid >> 2) << 11) + kb + (tid & 3) * 8;
        async_copy_b128(&vbuf[buf][tid * 8], g);
    };

    const float slope = exp2f(-0.5f * (float)(h + 1));  // 2^(-8(h+1)/16)
    const float inv_scale = 0.125f;                     // 1/sqrt(64)

    stageK(0, 0);
    stageV(0, 0);

    // Q A-fragments (d = 0..31 and 32..63), reused for all key blocks.
    Frag qa0, qa1;
    {
        const unsigned short* qr = Qs + (size_t)(q0 + ln) * 64;
        qa0.q[0] = *(const uint4*)(qr + half * 8);
        qa0.q[1] = *(const uint4*)(qr + 16 + half * 8);
        qa1.q[0] = *(const uint4*)(qr + 32 + half * 8);
        qa1.q[1] = *(const uint4*)(qr + 48 + half * 8);
    }

    const v8f zero = {0.f, 0.f, 0.f, 0.f, 0.f, 0.f, 0.f, 0.f};
    v8f acc[4] = {zero, zero, zero, zero};
    float mprev[8], lsum[8];
#pragma unroll
    for (int i = 0; i < 8; ++i) { mprev[i] = -1e30f; lsum[i] = 0.f; }

    int buf = 0;
    for (int kb = 0; kb < 2048; kb += 32) {
        wait_async0();
        __syncthreads();

        if (kb + 32 < 2048) { stageK(kb + 32, buf ^ 1); stageV(kb + 32, buf ^ 1); }

        // ---- scores: S[16q x 32k] as two 16x16 tiles, 2 WMMAs each --------
        v8f s[2] = {zero, zero};
#pragma unroll
        for (int t = 0; t < 2; ++t) {
            Frag bk0, bk1;   // B: col = key pos (row t*16+ln of tile), K = d
            const uint4* p0 = (const uint4*)&kbuf[buf][(t * 16 + ln) * 64 + half * 8];
            const uint4* p1 = (const uint4*)&kbuf[buf][(t * 16 + ln) * 64 + 32 + half * 8];
            bk0.q[0] = p0[0]; bk0.q[1] = p0[2];
            bk1.q[0] = p1[0]; bk1.q[1] = p1[2];
            s[t] = wmma_bf16(qa0.b, bk0.b, s[t]);
            s[t] = wmma_bf16(qa1.b, bk1.b, s[t]);
        }

        // ---- scale + ALiBi + online softmax -------------------------------
        float sm[2][8];
#pragma unroll
        for (int t = 0; t < 2; ++t)
#pragma unroll
            for (int i = 0; i < 8; ++i) {
                const int qpos = q0 + i + 8 * half;
                const int kpos = kb + t * 16 + ln;
                sm[t][i] = s[t][i] * inv_scale + slope * (float)(kpos - qpos);
            }

#pragma unroll
        for (int i = 0; i < 8; ++i) {
            float rm = fmaxf(sm[0][i], sm[1][i]);
#pragma unroll
            for (int d = 1; d < 16; d <<= 1) rm = fmaxf(rm, __shfl_xor(rm, d, 32));
            const float mn   = fmaxf(mprev[i], rm);
            const float corr = __expf(mprev[i] - mn);
            sm[0][i] = __expf(sm[0][i] - mn);
            sm[1][i] = __expf(sm[1][i] - mn);
            float rs = sm[0][i] + sm[1][i];
#pragma unroll
            for (int d = 1; d < 16; d <<= 1) rs += __shfl_xor(rs, d, 32);
            lsum[i]  = lsum[i] * corr + rs;
            mprev[i] = mn;
#pragma unroll
            for (int t = 0; t < 4; ++t) acc[t][i] *= corr;
        }

        // ---- re-shape P (C layout f32) -> A layout bf16 via LDS -----------
#pragma unroll
        for (int t = 0; t < 2; ++t)
#pragma unroll
            for (int i = 0; i < 8; ++i)
                pbuf[wv][(i + 8 * half) * 32 + t * 16 + ln] = f2bf(sm[t][i]);

        asm volatile("s_wait_dscnt 0" ::: "memory");   // wave-local LDS RAW

        Frag pa;
        {
            const unsigned short* pr = &pbuf[wv][ln * 32 + half * 16];
            pa.q[0] = *(const uint4*)(pr);
            pa.q[1] = *(const uint4*)(pr + 8);
        }

        // ---- O += P (16x32) * V^T-tiles (32k x 16d), 4 WMMAs --------------
        Frag vb[4];
#pragma unroll
        for (int t = 0; t < 4; ++t) {
            const uint4* vp = (const uint4*)&vbuf[buf][(t * 16 + ln) * 32 + half * 16];
            vb[t].q[0] = vp[0];
            vb[t].q[1] = vp[1];
        }
#pragma unroll
        for (int t = 0; t < 4; ++t) acc[t] = wmma_bf16(pa.b, vb[t].b, acc[t]);

        buf ^= 1;
    }

    // ---- normalize + store ctx[B*Sq, 1024] (col = h*64 + d) ---------------
#pragma unroll
    for (int i = 0; i < 8; ++i) {
        const float inv = 1.0f / lsum[i];
        const int m = q0 + i + 8 * half;
        const size_t row = ((size_t)(b << 11) + m) << 10;   // *1024
#pragma unroll
        for (int t = 0; t < 4; ++t)
            ctx[row + h * 64 + t * 16 + ln] = f2bf(acc[t][i] * inv);
    }
}

// ---------------------------------------------------------------------------
extern "C" void kernel_launch(void* const* d_in, const int* in_sizes, int n_in,
                              void* d_out, int out_size, void* d_ws, size_t ws_size,
                              hipStream_t stream) {
    const float* q   = (const float*)d_in[0];
    const float* k   = (const float*)d_in[1];
    const float* v   = (const float*)d_in[2];
    const float* w_q = (const float*)d_in[3];
    const float* w_k = (const float*)d_in[4];
    const float* w_v = (const float*)d_in[5];
    const float* w_o = (const float*)d_in[6];
    const float* b_o = (const float*)d_in[7];

    const size_t M1 = (size_t)1 << 20;         // 1M elements
    unsigned short* ws = (unsigned short*)d_ws;
    unsigned short* qx  = ws;                  // [4096,1024] bf16
    unsigned short* kx  = ws + 4 * M1;
    unsigned short* vx  = ws + 8 * M1;
    unsigned short* wqb = ws + 12 * M1;        // [1024,1024] bf16
    unsigned short* wkb = ws + 13 * M1;
    unsigned short* wvb = ws + 14 * M1;
    unsigned short* wob = ws + 15 * M1;
    unsigned short* Qh  = ws + 16 * M1;        // [B,H,2048,64]
    unsigned short* Kh  = ws + 20 * M1;
    unsigned short* Vtb = ws + 24 * M1;        // [B,H,64,2048]
    unsigned short* ctx = ws + 28 * M1;        // [4096,1024]

    const int nAct = 2 * 2048 * 1024;          // 4M
    const int nW   = 1024 * 1024;              // 1M
    cvt_bf16_kernel<<<(nAct + 255) / 256, 256, 0, stream>>>(q,   qx,  nAct);
    cvt_bf16_kernel<<<(nAct + 255) / 256, 256, 0, stream>>>(k,   kx,  nAct);
    cvt_bf16_kernel<<<(nAct + 255) / 256, 256, 0, stream>>>(v,   vx,  nAct);
    cvt_bf16_kernel<<<(nW   + 255) / 256, 256, 0, stream>>>(w_q, wqb, nW);
    cvt_bf16_kernel<<<(nW   + 255) / 256, 256, 0, stream>>>(w_k, wkb, nW);
    cvt_bf16_kernel<<<(nW   + 255) / 256, 256, 0, stream>>>(w_v, wvb, nW);
    cvt_bf16_kernel<<<(nW   + 255) / 256, 256, 0, stream>>>(w_o, wob, nW);

    // QKV projections (M=4096, N=1024, K=1024); block tile 128x64
    dim3 ggrid(4096 / 128, 1024 / 64);
    gemm_xwt_kernel<<<ggrid, 256, 0, stream>>>(qx, wqb, Qh,  nullptr, nullptr,
                                               4096, 1024, 1024, 0);
    gemm_xwt_kernel<<<ggrid, 256, 0, stream>>>(kx, wkb, Kh,  nullptr, nullptr,
                                               4096, 1024, 1024, 1);
    gemm_xwt_kernel<<<ggrid, 256, 0, stream>>>(vx, wvb, Vtb, nullptr, nullptr,
                                               4096, 1024, 1024, 2);

    // Attention: 512 blocks x 8 waves; block = one (b,h,q-group)
    attn_alibi_kernel<<<512, 256, 0, stream>>>(Qh, Kh, Vtb, ctx);

    // Output projection with bias -> fp32 d_out
    gemm_xwt_kernel<<<ggrid, 256, 0, stream>>>(ctx, wob, nullptr, (float*)d_out,
                                               b_o, 4096, 1024, 1024, 3);
}
